// probabilistic_MIL_Bayes_spvis_simplify_47012712022229
// MI455X (gfx1250) — matrix operand order repaired and checked by reading, not compiled, not run on hardware
//
#include <hip/hip_runtime.h>

// ---------------------------------------------------------------------------
// Types / feature detection for CDNA5
// ---------------------------------------------------------------------------
typedef __attribute__((ext_vector_type(16))) __bf16     v16bf;
typedef __attribute__((ext_vector_type(2)))  __bf16     v2bf;
typedef __attribute__((ext_vector_type(8)))  float      v8f;
typedef __attribute__((ext_vector_type(4)))  unsigned   u32x4;
typedef __attribute__((ext_vector_type(8)))  unsigned   u32x8;
typedef __attribute__((ext_vector_type(4)))  int        v4i;

#define GLOBAL_AS __attribute__((address_space(1)))
#define LDS_AS    __attribute__((address_space(3)))

#if defined(__has_builtin)
#if __has_builtin(__builtin_amdgcn_global_load_async_to_lds_b128)
#define HAVE_ASYNC_LDS 1
#endif
#endif
#ifndef HAVE_ASYNC_LDS
#define HAVE_ASYNC_LDS 0
#endif

#if defined(__has_builtin) && __has_builtin(__builtin_amdgcn_s_wait_asynccnt)
#define ASYNC_WAIT() __builtin_amdgcn_s_wait_asynccnt(0)
#else
#define ASYNC_WAIT() asm volatile("s_wait_asynccnt 0" ::: "memory")
#endif

#define HCC 256
#define WCC 256
#define GRID_CELLS (HCC * WCC)

__device__ __forceinline__ unsigned short f32_to_bf16(float f) {
  unsigned u = __float_as_uint(f);
  unsigned r = u + 0x7fffu + ((u >> 16) & 1u);   // round-to-nearest-even
  return (unsigned short)(r >> 16);
}

__device__ __forceinline__ unsigned pack_bf16x2(float lo, float hi) {
#if defined(__has_builtin) && __has_builtin(__builtin_amdgcn_cvt_pk_bf16_f32)
  v2bf p = __builtin_amdgcn_cvt_pk_bf16_f32(lo, hi);
  return __builtin_bit_cast(unsigned, p);
#else
  return ((unsigned)f32_to_bf16(hi) << 16) | f32_to_bf16(lo);
#endif
}

// ---------------------------------------------------------------------------
// f32 -> bf16 conversion for the small weight matrices only
// ---------------------------------------------------------------------------
__global__ void k_cvt_bf16(const float* __restrict__ src,
                           unsigned short* __restrict__ dst, long n) {
  long i = (long)blockIdx.x * blockDim.x + threadIdx.x;
  long stride = (long)gridDim.x * blockDim.x;
  for (; i < n; i += stride) dst[i] = f32_to_bf16(src[i]);
}

// ---------------------------------------------------------------------------
// GEMM1: x = relu(h @ W1^T + b1)   [N,1024]x[1024,512] -> [N,512]
// Block = 256 thr (8 waves). Block tile: 32 rows x 512 cols. Wave: 2 row-tiles
// x 4 col-tiles = 8 independent WMMA chains; each B fragment feeds 2 row-tiles
// (halved L2 traffic). h is read ONCE as f32 and converted to bf16 during LDS
// staging; K staged 64 at a time, register double-buffered.
// ---------------------------------------------------------------------------
__global__ void k_gemm1(const float* __restrict__ h,
                        const unsigned short* __restrict__ w1bf,
                        const float* __restrict__ b1,
                        float* __restrict__ x,
                        unsigned short* __restrict__ xbf, int N) {
  __shared__ __align__(16) unsigned short As[32 * 64];   // K=64 chunk, 4KB
  const int lane = threadIdx.x & 31;
  const int wave = threadIdx.x >> 5;
  const int ll = lane & 15, lh = lane >> 4;
  const int r0 = blockIdx.x * 32;
  const int c0 = wave * 64;                    // 4 col tiles per wave

  v8f zero = {};
  v8f acc[2][4];
#pragma unroll
  for (int rt = 0; rt < 2; ++rt)
#pragma unroll
    for (int ct = 0; ct < 4; ++ct) acc[rt][ct] = zero;

  const int srow = threadIdx.x >> 3;           // staging row 0..31
  const int sseg = (threadIdx.x & 7) * 8;      // f32 col within chunk (8/thread)
  int rclamp = r0 + srow; if (rclamp >= N) rclamp = N - 1;
  const float* gsrc = h + (size_t)rclamp * 1024 + sseg;

  float4 pre0 = *(const float4*)gsrc;          // prefetch chunk kb=0
  float4 pre1 = *(const float4*)(gsrc + 4);
  for (int kb = 0; kb < 1024; kb += 64) {
    uint4 pk;
    pk.x = pack_bf16x2(pre0.x, pre0.y);
    pk.y = pack_bf16x2(pre0.z, pre0.w);
    pk.z = pack_bf16x2(pre1.x, pre1.y);
    pk.w = pack_bf16x2(pre1.z, pre1.w);
    *(uint4*)((unsigned*)As + srow * 32 + (sseg >> 1)) = pk;
    __syncthreads();
    if (kb + 64 < 1024) {                      // overlap next load with WMMA
      pre0 = *(const float4*)(gsrc + kb + 64);
      pre1 = *(const float4*)(gsrc + kb + 68);
    }

#pragma unroll
    for (int ks = 0; ks < 64; ks += 32) {
      v16bf a[2];
#pragma unroll
      for (int rt = 0; rt < 2; ++rt) {
        const u32x4* ap = (const u32x4*)(As + (rt * 16 + ll) * 64 + ks + lh * 16);
        u32x4 a0 = ap[0], a1 = ap[1];
        u32x8 av = {a0.x, a0.y, a0.z, a0.w, a1.x, a1.y, a1.z, a1.w};
        a[rt] = __builtin_bit_cast(v16bf, av);
      }
#pragma unroll
      for (int ct = 0; ct < 4; ++ct) {
        int nB = c0 + ct * 16 + ll;
        v16bf b = __builtin_bit_cast(v16bf,
            *(const u32x8*)(w1bf + (size_t)nB * 1024 + kb + ks + lh * 16));
        acc[0][ct] = __builtin_amdgcn_wmma_f32_16x16x32_bf16(
            false, a[0], false, b, (short)0, acc[0][ct], false, false);
        acc[1][ct] = __builtin_amdgcn_wmma_f32_16x16x32_bf16(
            false, a[1], false, b, (short)0, acc[1][ct], false, false);
      }
    }
    __syncthreads();
  }

#pragma unroll
  for (int ct = 0; ct < 4; ++ct) {
    int nB = c0 + ct * 16 + ll;
    float bias = b1[nB];
#pragma unroll
    for (int rt = 0; rt < 2; ++rt) {
#pragma unroll
      for (int i = 0; i < 8; ++i) {
        int m = r0 + rt * 16 + i + lh * 8;
        if (m < N) {
          float v = fmaxf(acc[rt][ct][i] + bias, 0.0f);
          x[(size_t)m * 512 + nB]   = v;
          xbf[(size_t)m * 512 + nB] = f32_to_bf16(v);
        }
      }
    }
  }
}

// ---------------------------------------------------------------------------
// GEMM2 (fused): fa = sigmoid(x@W2a^T+b2a), fb = tanh(x@W2b^T+b2b),
// g = fa*fb -> LDS, then params = g @ W3^T + b3  -> [N,2]
// Block tile: 32 rows x 256 cols; wave: 2 row-tiles x (2 cols x 2 mats) = 8
// WMMA chains, B fragments shared across row-tiles. A tile staged with
// GLOBAL_LOAD_ASYNC_TO_LDS_B128 (double-buffered, ASYNCcnt) when available.
// ---------------------------------------------------------------------------
__global__ void k_gemm2(const unsigned short* __restrict__ xbf,
                        const unsigned short* __restrict__ w2abf,
                        const unsigned short* __restrict__ w2bbf,
                        const float* __restrict__ b2a, const float* __restrict__ b2b,
                        const float* __restrict__ W3, const float* __restrict__ b3,
                        float* __restrict__ params, int N) {
#if HAVE_ASYNC_LDS
  __shared__ __align__(16) unsigned short As[2][32 * 64];   // 8KB
#else
  __shared__ __align__(16) unsigned short As[1][32 * 64];
#endif
  __shared__ float gbuf[32 * 256];   // 32KB
  __shared__ float red[256];

  const int lane = threadIdx.x & 31;
  const int wave = threadIdx.x >> 5;
  const int ll = lane & 15, lh = lane >> 4;
  const int r0 = blockIdx.x * 32;
  const int n0 = wave * 32 + ll, n1 = n0 + 16;

  v8f zero = {};
  v8f aa[2][2], ab[2][2];            // [row-tile][col-tile]
#pragma unroll
  for (int rt = 0; rt < 2; ++rt)
#pragma unroll
    for (int ct = 0; ct < 2; ++ct) { aa[rt][ct] = zero; ab[rt][ct] = zero; }

  auto compute_chunk = [&](const unsigned short* Ab, int kb) {
#pragma unroll
    for (int ks = 0; ks < 64; ks += 32) {
      v16bf a[2];
#pragma unroll
      for (int rt = 0; rt < 2; ++rt) {
        const u32x4* ap = (const u32x4*)(Ab + (rt * 16 + ll) * 64 + ks + lh * 16);
        u32x4 a0 = ap[0], a1 = ap[1];
        u32x8 av = {a0.x, a0.y, a0.z, a0.w, a1.x, a1.y, a1.z, a1.w};
        a[rt] = __builtin_bit_cast(v16bf, av);
      }
      v16bf ba0 = __builtin_bit_cast(v16bf,
          *(const u32x8*)(w2abf + (size_t)n0 * 512 + kb + ks + lh * 16));
      v16bf ba1 = __builtin_bit_cast(v16bf,
          *(const u32x8*)(w2abf + (size_t)n1 * 512 + kb + ks + lh * 16));
      v16bf bb0 = __builtin_bit_cast(v16bf,
          *(const u32x8*)(w2bbf + (size_t)n0 * 512 + kb + ks + lh * 16));
      v16bf bb1 = __builtin_bit_cast(v16bf,
          *(const u32x8*)(w2bbf + (size_t)n1 * 512 + kb + ks + lh * 16));
#pragma unroll
      for (int rt = 0; rt < 2; ++rt) {
        aa[rt][0] = __builtin_amdgcn_wmma_f32_16x16x32_bf16(false, a[rt], false, ba0, (short)0, aa[rt][0], false, false);
        aa[rt][1] = __builtin_amdgcn_wmma_f32_16x16x32_bf16(false, a[rt], false, ba1, (short)0, aa[rt][1], false, false);
        ab[rt][0] = __builtin_amdgcn_wmma_f32_16x16x32_bf16(false, a[rt], false, bb0, (short)0, ab[rt][0], false, false);
        ab[rt][1] = __builtin_amdgcn_wmma_f32_16x16x32_bf16(false, a[rt], false, bb1, (short)0, ab[rt][1], false, false);
      }
    }
  };

  const int arow = threadIdx.x >> 3;            // staging row 0..31
  const int aseg = (threadIdx.x & 7) * 8;       // bf16 col (16B per thread)
  int arc = r0 + arow; if (arc >= N) arc = N - 1;
  const unsigned short* agsrc = xbf + (size_t)arc * 512 + aseg;

#if HAVE_ASYNC_LDS
  auto issue_chunk = [&](int buf, int kb) {
    __builtin_amdgcn_global_load_async_to_lds_b128(
        (GLOBAL_AS v4i*)(agsrc + kb),
        (LDS_AS v4i*)(&As[buf][arow * 64 + aseg]), 0, 0);
  };
  issue_chunk(0, 0);
  ASYNC_WAIT();
  __syncthreads();
  for (int i = 0; i < 8; ++i) {
    int kb = i * 64;
    if (i + 1 < 8) issue_chunk((i + 1) & 1, kb + 64);  // overlap copy w/ WMMA
    compute_chunk(As[i & 1], kb);
    __syncthreads();                                    // all reads of buf done
    if (i + 1 < 8) { ASYNC_WAIT(); __syncthreads(); }   // next chunk landed
  }
#else
  uint4 pre = *(const uint4*)agsrc;
  for (int i = 0; i < 8; ++i) {
    int kb = i * 64;
    *(uint4*)((unsigned*)As[0] + arow * 32 + (aseg >> 1)) = pre;
    __syncthreads();
    if (kb + 64 < 512) pre = *(const uint4*)(agsrc + kb + 64);
    compute_chunk(As[0], kb);
    __syncthreads();
  }
#endif

  const float bia0 = b2a[n0], bia1 = b2a[n1], bib0 = b2b[n0], bib1 = b2b[n1];
#pragma unroll
  for (int rt = 0; rt < 2; ++rt) {
#pragma unroll
    for (int i = 0; i < 8; ++i) {
      int mrow = rt * 16 + i + lh * 8;
      float fa0 = 1.0f / (1.0f + expf(-(aa[rt][0][i] + bia0)));
      float fb0 = tanhf(ab[rt][0][i] + bib0);
      float fa1 = 1.0f / (1.0f + expf(-(aa[rt][1][i] + bia1)));
      float fb1 = tanhf(ab[rt][1][i] + bib1);
      gbuf[mrow * 256 + n0] = fa0 * fb0;
      gbuf[mrow * 256 + n1] = fa1 * fb1;
    }
  }
  __syncthreads();

  // params[m][j] = sum_n g[m][n]*W3[j][n] + b3[j]; 64 (row,j) pairs x 4 lanes
  {
    int p = threadIdx.x >> 2;      // 0..63 -> (row = p>>1, j = p&1)
    int row = p >> 1, j = p & 1;
    int s = threadIdx.x & 3;
    float part = 0.f;
    for (int q = s; q < 256; q += 4) part += gbuf[row * 256 + q] * W3[j * 256 + q];
    red[threadIdx.x] = part;
  }
  __syncthreads();
  if (threadIdx.x < 64) {
    int t = threadIdx.x;
    float tot = red[t * 4] + red[t * 4 + 1] + red[t * 4 + 2] + red[t * 4 + 3];
    int row = t >> 1, j = t & 1;
    int m = r0 + row;
    if (m < N) params[(size_t)m * 2 + j] = tot + b3[j];
  }
}

// ---------------------------------------------------------------------------
__global__ void k_zero(float* mu, float* lv, float* msum, float* S) {
  int i = blockIdx.x * 256 + threadIdx.x;
  if (i < GRID_CELLS) { mu[i] = 0.f; lv[i] = 0.f; }
  if (i < 512) msum[i] = 0.f;
  if (i == 0) S[0] = 0.f;
}

__global__ void k_scatter(const int* __restrict__ coords,
                          const float* __restrict__ params,
                          float* __restrict__ mu, float* __restrict__ lv, int N) {
  int i = blockIdx.x * 256 + threadIdx.x;
  if (i >= N) return;
  int gx = coords[2 * i] >> 8;
  int gy = coords[2 * i + 1] >> 8;
  int c = gy * WCC + gx;
  mu[c] = params[2 * i];
  lv[c] = params[2 * i + 1];
}

__global__ void k_klconv(const float* __restrict__ mu, const float* __restrict__ lv,
                         const float* __restrict__ eps, const float* __restrict__ kern,
                         const int* __restrict__ slide_label,
                         float* __restrict__ Agrid, float* __restrict__ out_kl) {
  int c = blockIdx.x * 256 + threadIdx.x;
  if (c >= GRID_CELLS) return;
  int gy = c >> 8, gx = c & 255;
  int lbl = slide_label[0];
  float pmu = (lbl == 0) ? -5.0f : 0.0f;
  float plv = (lbl == 0) ? -1.0f : 3.0f;
  float m = mu[c], l = lv[c];
  out_kl[c] = (plv - l) * 0.5f + (l * l + (pmu - m) * (pmu - m)) / (2.0f * plv * plv) - 0.5f;

  float s = 0.f;
#pragma unroll
  for (int dy = -1; dy <= 1; ++dy)
#pragma unroll
    for (int dx = -1; dx <= 1; ++dx) {
      int yy = gy + dy, xx = gx + dx;
      float v = (yy >= 0 && yy < HCC && xx >= 0 && xx < WCC) ? mu[yy * WCC + xx] : 0.f;
      s += v * kern[(dy + 1) * 3 + (dx + 1)];
    }
  float sample = s + eps[c] * expf(0.5f * l);
  Agrid[c] = 1.0f / (1.0f + expf(-sample));
}

// ---------------------------------------------------------------------------
// Gather patch_A, Σa, Msum[512] = Σ_i a_i*x[i][:]; thread t owns cols 2t,2t+1
// ---------------------------------------------------------------------------
__global__ void k_gather(const int* __restrict__ coords, const float* __restrict__ Agrid,
                         const float* __restrict__ x, float* __restrict__ patchA_out,
                         float* __restrict__ msum, float* __restrict__ S, int N) {
  __shared__ float a_sh[256];
  __shared__ float s_red[256];
  int base = blockIdx.x * 256;
  int i = base + threadIdx.x;
  float a = 0.f;
  if (i < N) {
    int gx = coords[2 * i] >> 8, gy = coords[2 * i + 1] >> 8;
    a = Agrid[gy * WCC + gx];
    patchA_out[i] = a;
  }
  a_sh[threadIdx.x] = a;
  s_red[threadIdx.x] = a;
  __syncthreads();
  for (int st = 128; st > 0; st >>= 1) {
    if (threadIdx.x < st) s_red[threadIdx.x] += s_red[threadIdx.x + st];
    __syncthreads();
  }
  if (threadIdx.x == 0) atomicAdd(S, s_red[0]);

  int nrows = N - base; if (nrows > 256) nrows = 256;
  float acc0 = 0.f, acc1 = 0.f;
  for (int r = 0; r < nrows; ++r) {
    const float2* xr = (const float2*)(x + (size_t)(base + r) * 512);
    if (r + 4 < nrows)
      __builtin_prefetch(x + (size_t)(base + r + 4) * 512 + 2 * threadIdx.x, 0, 0);
    float ar = a_sh[r];
    float2 v = xr[threadIdx.x];
    acc0 += ar * v.x;
    acc1 += ar * v.y;
  }
  atomicAdd(&msum[2 * threadIdx.x], acc0);
  atomicAdd(&msum[2 * threadIdx.x + 1], acc1);
}

// ---------------------------------------------------------------------------
__global__ void k_final(const float* __restrict__ msum, const float* __restrict__ S,
                        const float* __restrict__ Wc, const float* __restrict__ bc,
                        float* __restrict__ out) {
  __shared__ float red[128];
  int t = threadIdx.x;  // 64 threads
  float inv = 1.0f / S[0];
  float p0 = 0.f, p1 = 0.f;
  for (int c = t; c < 512; c += 64) {
    float m = msum[c] * inv;
    p0 += m * Wc[c];
    p1 += m * Wc[512 + c];
  }
  red[t] = p0; red[64 + t] = p1;
  __syncthreads();
  for (int st = 32; st > 0; st >>= 1) {
    if (t < st) { red[t] += red[t + st]; red[64 + t] += red[64 + t + st]; }
    __syncthreads();
  }
  if (t == 0) {
    float l0 = red[0] + bc[0], l1 = red[64] + bc[1];
    float mx = fmaxf(l0, l1);
    float e0 = expf(l0 - mx), e1 = expf(l1 - mx);
    float den = e0 + e1;
    float q0 = e0 / den, q1 = e1 / den;
    out[0] = l0; out[1] = l1;                    // top_instance
    out[2] = q0; out[3] = q1;                    // Y_prob
    out[4] = (l1 > l0) ? 1.0f : 0.0f;            // Y_hat
    out[5 + GRID_CELLS]     = q0;                // y_probs
    out[5 + GRID_CELLS + 1] = q1;
  }
}

// ---------------------------------------------------------------------------
extern "C" void kernel_launch(void* const* d_in, const int* in_sizes, int n_in,
                              void* d_out, int out_size, void* d_ws, size_t ws_size,
                              hipStream_t stream) {
  const float* h        = (const float*)d_in[0];
  const int*   coords   = (const int*)d_in[1];
  const float* eps      = (const float*)d_in[2];
  const float* W1       = (const float*)d_in[3];
  const float* b1       = (const float*)d_in[4];
  const float* W2a      = (const float*)d_in[5];
  const float* b2a      = (const float*)d_in[6];
  const float* W2b      = (const float*)d_in[7];
  const float* b2b      = (const float*)d_in[8];
  const float* W3       = (const float*)d_in[9];
  const float* b3       = (const float*)d_in[10];
  const float* Wc       = (const float*)d_in[11];
  const float* bc       = (const float*)d_in[12];
  const float* kern     = (const float*)d_in[13];
  const int*   slide_lb = (const int*)d_in[16];

  const int N = in_sizes[0] / 1024;
  float* out = (float*)d_out;

  char* ws = (char*)d_ws;
  size_t off = 0;
  auto alloc = [&](size_t bytes) -> char* {
    char* p = ws + off;
    off = (off + bytes + 255) & ~(size_t)255;
    return p;
  };
  unsigned short* w1bf   = (unsigned short*)alloc((size_t)512 * 1024 * 2);
  unsigned short* w2abf  = (unsigned short*)alloc((size_t)256 * 512 * 2);
  unsigned short* w2bbf  = (unsigned short*)alloc((size_t)256 * 512 * 2);
  float*          xbuf   = (float*)alloc((size_t)N * 512 * 4);
  unsigned short* xbf    = (unsigned short*)alloc((size_t)N * 512 * 2);
  float*          params = (float*)alloc((size_t)N * 2 * 4);
  float*          mu     = (float*)alloc((size_t)GRID_CELLS * 4);
  float*          lv     = (float*)alloc((size_t)GRID_CELLS * 4);
  float*          Agrid  = (float*)alloc((size_t)GRID_CELLS * 4);
  float*          msum   = (float*)alloc(512 * 4);
  float*          Ssum   = (float*)alloc(256);

  const int mt32 = (N + 31) / 32;

  k_cvt_bf16<<<512, 256, 0, stream>>>(W1,  w1bf,  (long)512 * 1024);
  k_cvt_bf16<<<128, 256, 0, stream>>>(W2a, w2abf, (long)256 * 512);
  k_cvt_bf16<<<128, 256, 0, stream>>>(W2b, w2bbf, (long)256 * 512);

  k_gemm1<<<mt32, 256, 0, stream>>>(h, w1bf, b1, xbuf, xbf, N);
  k_gemm2<<<mt32, 256, 0, stream>>>(xbf, w2abf, w2bbf, b2a, b2b, W3, b3, params, N);

  k_zero<<<GRID_CELLS / 256, 256, 0, stream>>>(mu, lv, msum, Ssum);
  k_scatter<<<(N + 255) / 256, 256, 0, stream>>>(coords, params, mu, lv, N);
  k_klconv<<<GRID_CELLS / 256, 256, 0, stream>>>(mu, lv, eps, kern, slide_lb,
                                                 Agrid, out + 5);
  k_gather<<<(N + 255) / 256, 256, 0, stream>>>(coords, Agrid, xbuf,
                                                out + 5 + GRID_CELLS + 2, msum, Ssum, N);
  k_final<<<1, 64, 0, stream>>>(msum, Ssum, Wc, bc, out);
}
